// MPNNPOM_3839700762684
// MI455X (gfx1250) — matrix operand order repaired and checked by reading, not compiled, not run on hardware
//
#include <hip/hip_runtime.h>
#include <hip/hip_bf16.h>
#include <cmath>

// ---------------------------------------------------------------------------
// MPNNPOM forward for MI455X (gfx1250, wave32, WMMA).
// All dense GEMMs use v_wmma_f32_16x16x32_f16 with f16 operands end-to-end:
// weights are pre-converted once per launch into f16 N-major (B[n*K+k]) so
// every fragment load is a contiguous 16-byte global_load_b128; activations
// feeding GEMMs are produced as f16 mirrors. Hot GEMMs (edge_net2 42 GFLOP,
// GRU, LSTM) take the guard-free path: 6x b128 loads + 2x wmma per K-step.
// `ew` (E x 64 x 64) is stored f16 (327 MB) to halve HBM traffic for the 3
// message rounds (~1 GB total @ 23.3 TB/s ~= 45 us, the roofline floor).
// Workspace requirement: ~430 MB.
// ---------------------------------------------------------------------------

typedef __attribute__((ext_vector_type(16))) _Float16 v16h;
typedef __attribute__((ext_vector_type(8)))  _Float16 h8;
typedef __attribute__((ext_vector_type(2)))  _Float16 h2v;
typedef __attribute__((ext_vector_type(8)))  float    v8f;

static constexpr int Vn   = 20000;   // nodes   (multiple of 16)
static constexpr int En   = 40000;   // edges   (multiple of 16)
static constexpr int Bg   = 512;     // graphs  (multiple of 16)
static constexpr int NATF = 134;
static constexpr int NBF  = 6;
static constexpr int DN   = 64;
static constexpr int DEH  = 128;
static constexpr int DS2S = 128;
static constexpr int FFNIN  = 256;
static constexpr int FFNHID = 300;
static constexpr int FFNEMB = 256;
static constexpr int NTASK  = 138;
static constexpr int NSTEPS = 3;
static constexpr int NS2S   = 6;
static constexpr int NLSTM  = 3;

__device__ __forceinline__ float sigf(float x) { return 1.f / (1.f + __expf(-x)); }

// ---------------------------------------------------------------------------
// WMMA GEMM: Out[M,N] = act(A[M,K](f16) * Bt[N,K](f16,N-major) + bias)
// One wave owns two adjacent 16x16 output tiles (shared A fragment).
// M must be a multiple of 16 (true for all call sites).
// TAIL=false requires K%32==0 && N%32==0: pure b128 loads, no guards.
// TAIL=true : clamped-address scalar loads + select (no EXEC divergence).
// Fragment layouts per CDNA5 ISA 7.12.2:
//   A lane: M=lane&15; k-halves [8*hi,8*hi+8) and [16+8*hi, 16+8*hi+8)
//   B lane: N=lane&15; k-halves [8*hi,8*hi+8) and [16+8*hi, 16+8*hi+8)
//   D VGPR r: M = r + 8*hi, N = lane&15
// ---------------------------------------------------------------------------
template<bool RELU, bool HOUT, bool TAIL>
__global__ __launch_bounds__(128)
void k_gemm(const _Float16* __restrict__ A, const _Float16* __restrict__ Bt,
            const float* __restrict__ bias, void* __restrict__ Out,
            int M, int N, int K)
{
    const int lane = threadIdx.x & 31;
    const int wave = threadIdx.x >> 5;
    const int tm   = blockIdx.x << 4;
    const int tn   = ((blockIdx.y << 2) + wave) << 5;   // 2 n-tiles per wave
    if (tn >= N) return;                                 // whole-wave exit
    const int hi  = lane >> 4;
    const int l15 = lane & 15;
    const int am  = tm + l15;
    const int bn0 = tn + l15;
    const int bn1 = tn + 16 + l15;

    v8f acc0 = {}, acc1 = {};
    for (int kk = 0; kk < K; kk += 32) {
        v16h a, b0, b1;
        if (!TAIL) {
            const _Float16* ap = A + (size_t)am * K + kk + (hi << 3);
            const h8 al = *(const h8*)ap;
            const h8 ah = *(const h8*)(ap + 16);
            a = __builtin_shufflevector(al, ah, 0,1,2,3,4,5,6,7,8,9,10,11,12,13,14,15);
            const _Float16* bp0 = Bt + (size_t)bn0 * K + kk + (hi << 3);
            b0 = __builtin_shufflevector(*(const h8*)bp0, *(const h8*)(bp0 + 16),
                                         0,1,2,3,4,5,6,7,8,9,10,11,12,13,14,15);
            const _Float16* bp1 = Bt + (size_t)bn1 * K + kk + (hi << 3);
            b1 = __builtin_shufflevector(*(const h8*)bp1, *(const h8*)(bp1 + 16),
                                         0,1,2,3,4,5,6,7,8,9,10,11,12,13,14,15);
        } else {
            const int nc0 = min(bn0, N - 1), nc1 = min(bn1, N - 1);
#pragma unroll
            for (int i = 0; i < 16; ++i) {
                // A: i=2v+h -> k = 2*(v&3)+h + 16*(v>>2) + 8*hi
                const int vv = i >> 1, hh = i & 1;
                const int ka = kk + 2 * (vv & 3) + hh + ((vv >> 2) << 4) + (hi << 3);
                const int kac = min(ka, K - 1);
                const _Float16 xa = A[(size_t)am * K + kac];
                a[i] = (ka < K) ? xa : (_Float16)0.f;
                // B: k = (i&7) + 16*(i>>3) + 8*hi (columns past N hold garbage; never stored)
                const int kb = kk + (i & 7) + ((i >> 3) << 4) + (hi << 3);
                const int kbc = min(kb, K - 1);
                const _Float16 x0 = Bt[(size_t)nc0 * K + kbc];
                const _Float16 x1 = Bt[(size_t)nc1 * K + kbc];
                b0[i] = (kb < K) ? x0 : (_Float16)0.f;
                b1[i] = (kb < K) ? x1 : (_Float16)0.f;
            }
        }
        acc0 = __builtin_amdgcn_wmma_f32_16x16x32_f16(false, a, false, b0,
                                                      (short)0, acc0, false, false);
        acc1 = __builtin_amdgcn_wmma_f32_16x16x32_f16(false, a, false, b1,
                                                      (short)0, acc1, false, false);
    }

#pragma unroll
    for (int t = 0; t < 2; ++t) {
        const int n = tn + (t << 4) + l15;
        if (TAIL && n >= N) continue;
        const v8f& acc = t ? acc1 : acc0;
        const float bb = bias ? bias[n] : 0.f;
#pragma unroll
        for (int r = 0; r < 8; ++r) {
            const int m = tm + r + (hi << 3);
            float v = acc[r] + bb;
            if (RELU) v = fmaxf(v, 0.f);
            if (HOUT) ((_Float16*)Out)[(size_t)m * N + n] = (_Float16)v;
            else      ((float*)Out)[(size_t)m * N + n]    = v;
        }
    }
}

// ---------------------------------------------------------------------------
// Conversion / elementwise / scatter kernels
// ---------------------------------------------------------------------------
__global__ void k_zero(float* p, long n) {
    const long i = (long)blockIdx.x * blockDim.x + threadIdx.x;
    if (i < n) p[i] = 0.f;
}

__global__ void k_cvt(const float* __restrict__ in, _Float16* __restrict__ out, long n) {
    const long i = (long)blockIdx.x * blockDim.x + threadIdx.x;
    if (i < n) out[i] = (_Float16)in[i];
}

// (K,N) f32 row-major -> (N,K) f16 row-major (N-major weight prep)
__global__ void k_cvtT(const float* __restrict__ in, _Float16* __restrict__ out,
                       int K, int N) {
    const long i = (long)blockIdx.x * blockDim.x + threadIdx.x;
    if (i >= (long)K * N) return;
    const int k = (int)(i / N), n = (int)(i % N);
    out[(size_t)n * K + k] = (_Float16)in[i];
}

// msg[e] = h[src[e]] (1x64) @ ew[e] (64x64 f16), scatter-add into agg[dst[e]]
__global__ __launch_bounds__(256)
void k_msg_agg(const float* __restrict__ h, const _Float16* __restrict__ ew,
               const int* __restrict__ src, const int* __restrict__ dst,
               float* __restrict__ agg)
{
    const int e = blockIdx.x * 8 + (threadIdx.x >> 5);
    const int lane = threadIdx.x & 31;
    if (e >= En) return;
    const int s = src[e], d = dst[e];
    const _Float16* w = ew + (size_t)e * (DN * DN);
    float a0 = 0.f, a1 = 0.f;
#pragma unroll 4
    for (int i = 0; i < DN; ++i) {
        const float hv = h[(size_t)s * DN + i];
        const h2v wv = *(const h2v*)(w + i * DN + (lane << 1));  // o=2*lane,2*lane+1
        a0 += hv * (float)wv[0];
        a1 += hv * (float)wv[1];
    }
    atomicAdd(&agg[(size_t)d * DN + (lane << 1)],     a0);
    atomicAdd(&agg[(size_t)d * DN + (lane << 1) + 1], a1);
}

// m16 = f16( relu(agg + nnconv_b) + h )
__global__ void k_mix(const float* __restrict__ agg, const float* __restrict__ nnb,
                      const float* __restrict__ h, _Float16* __restrict__ m16)
{
    const int i = blockIdx.x * blockDim.x + threadIdx.x;
    if (i >= Vn * DN) return;
    m16[i] = (_Float16)(fmaxf(agg[i] + nnb[i & (DN - 1)], 0.f) + h[i]);
}

// GRU cell combine (torch gate order r,z,n); updates h (f32) and h16 mirror
__global__ void k_gru(const float* __restrict__ gi, const float* __restrict__ gh,
                      float* __restrict__ h, _Float16* __restrict__ h16)
{
    const int idx = blockIdx.x * blockDim.x + threadIdx.x;
    if (idx >= Vn * DN) return;
    const int v = idx >> 6, j = idx & 63;
    const size_t b = (size_t)v * (3 * DN);
    const float r = sigf(gi[b + j] + gh[b + j]);
    const float z = sigf(gi[b + DN + j] + gh[b + DN + j]);
    const float n = tanhf(gi[b + 2 * DN + j] + r * gh[b + 2 * DN + j]);
    const float hv = (1.f - z) * n + z * h[idx];
    h[idx]   = hv;
    h16[idx] = (_Float16)hv;
}

// feat[dst] += [h[src] ; e_proj[e]]
__global__ void k_feat(const float* __restrict__ h, const float* __restrict__ ep,
                       const int* __restrict__ src, const int* __restrict__ dst,
                       float* __restrict__ feat)
{
    const int idx = blockIdx.x * blockDim.x + threadIdx.x;
    if (idx >= En * 128) return;
    const int e = idx >> 7, j = idx & 127;
    const float v = (j < DN) ? h[(size_t)src[e] * DN + j]
                             : ep[(size_t)e * DN + (j - DN)];
    atomicAdd(&feat[(size_t)dst[e] * 128 + j], v);
}

// per-graph [start,end) from sorted graph_ids
__global__ void k_bounds(const int* __restrict__ gids, int* __restrict__ st,
                         int* __restrict__ en)
{
    const int v = blockIdx.x * blockDim.x + threadIdx.x;
    if (v >= Vn) return;
    const int g = gids[v];
    if (v == 0      || gids[v - 1] != g) st[g] = v;
    if (v == Vn - 1 || gids[v + 1] != g) en[g] = v + 1;
}

// LSTM cell combine (torch gate order i,f,g,o); biases already in ga/gb
__global__ void k_lstm(const float* __restrict__ ga, const float* __restrict__ gb,
                       float* __restrict__ hs, _Float16* __restrict__ hs16,
                       float* __restrict__ cs)
{
    const int idx = blockIdx.x * blockDim.x + threadIdx.x;
    if (idx >= Bg * DS2S) return;
    const int b = idx >> 7, j = idx & 127;
    const size_t o = (size_t)b * (4 * DS2S);
    const float gi_ = ga[o + j]       + gb[o + j];
    const float gf  = ga[o + 128 + j] + gb[o + 128 + j];
    const float gg  = ga[o + 256 + j] + gb[o + 256 + j];
    const float go  = ga[o + 384 + j] + gb[o + 384 + j];
    const float c = sigf(gf) * cs[idx] + sigf(gi_) * tanhf(gg);
    const float hv = sigf(go) * tanhf(c);
    cs[idx]   = c;
    hs[idx]   = hv;
    hs16[idx] = (_Float16)hv;
}

// e[v] = dot(feat[v], q[gid[v]]) — one wave per node
__global__ __launch_bounds__(256)
void k_escore(const float* __restrict__ feat, const float* __restrict__ q,
              const int* __restrict__ gids, float* __restrict__ esc)
{
    const int node = blockIdx.x * 8 + (threadIdx.x >> 5);
    const int lane = threadIdx.x & 31;
    if (node >= Vn) return;
    const int g = gids[node];
    float s = 0.f;
#pragma unroll
    for (int t = 0; t < 4; ++t) {
        const int k = lane + 32 * t;
        s += feat[(size_t)node * 128 + k] * q[(size_t)g * 128 + k];
    }
#pragma unroll
    for (int off = 16; off; off >>= 1) s += __shfl_xor(s, off, 32);
    if (lane == 0) esc[node] = s;
}

// per-graph softmax attention + readout; writes q_star16[g] = [q[g] ; readout]
__global__ __launch_bounds__(128)
void k_attn(const float* __restrict__ feat, const float* __restrict__ esc,
            const float* __restrict__ q, const int* __restrict__ st,
            const int* __restrict__ en, _Float16* __restrict__ qstar16)
{
    __shared__ float red[128];
    const int g = blockIdx.x, tid = threadIdx.x;
    const int s0 = st[g], s1 = en[g];
    float mx = -INFINITY;
    for (int v = s0 + tid; v < s1; v += 128) mx = fmaxf(mx, esc[v]);
    red[tid] = mx; __syncthreads();
    for (int s = 64; s; s >>= 1) { if (tid < s) red[tid] = fmaxf(red[tid], red[tid + s]); __syncthreads(); }
    const float emax = red[0]; __syncthreads();
    float sm = 0.f;
    for (int v = s0 + tid; v < s1; v += 128) sm += __expf(esc[v] - emax);
    red[tid] = sm; __syncthreads();
    for (int s = 64; s; s >>= 1) { if (tid < s) red[tid] += red[tid + s]; __syncthreads(); }
    const float denom = red[0]; __syncthreads();
    float r = 0.f;
    if (denom > 0.f) {
        for (int v = s0; v < s1; ++v) {
            const float w = __expf(esc[v] - emax) / denom;
            r += w * feat[(size_t)v * 128 + tid];
        }
    }
    qstar16[(size_t)g * 256 + tid]       = (_Float16)q[(size_t)g * 128 + tid];
    qstar16[(size_t)g * 256 + 128 + tid] = (_Float16)r;
}

__global__ void k_sigmoid(float* __restrict__ out) {
    const int i = blockIdx.x * blockDim.x + threadIdx.x;
    if (i >= Bg * NTASK) return;
    out[i] = sigf(out[Bg * NTASK + i]);   // proba = sigmoid(logits)
}

// ---------------------------------------------------------------------------
static inline int cdiv(int a, int b) { return (a + b - 1) / b; }

extern "C" void kernel_launch(void* const* d_in, const int* in_sizes, int n_in,
                              void* d_out, int out_size, void* d_ws, size_t ws_size,
                              hipStream_t stream)
{
    // ---- inputs (setup_inputs dict order, params flattened in insertion order)
    const float* node_feats = (const float*)d_in[0];
    const float* edge_feats = (const float*)d_in[1];
    const int*   src  = (const int*)d_in[2];
    const int*   dst  = (const int*)d_in[3];
    const int*   gids = (const int*)d_in[4];
    const float* Wpn = (const float*)d_in[5],  *bpn = (const float*)d_in[6];
    const float* We1 = (const float*)d_in[7],  *be1 = (const float*)d_in[8];
    const float* We2 = (const float*)d_in[9],  *be2 = (const float*)d_in[10];
    const float* nnb = (const float*)d_in[11];
    const float* gwih = (const float*)d_in[12], *gwhh = (const float*)d_in[13];
    const float* gbih = (const float*)d_in[14], *gbhh = (const float*)d_in[15];
    const float* Wpe = (const float*)d_in[16], *bpe = (const float*)d_in[17];
    const float *lwih[NLSTM], *lwhh[NLSTM], *lbih[NLSTM], *lbhh[NLSTM];
    for (int l = 0; l < NLSTM; ++l) {
        lwih[l] = (const float*)d_in[18 + 4 * l];
        lwhh[l] = (const float*)d_in[19 + 4 * l];
        lbih[l] = (const float*)d_in[20 + 4 * l];
        lbhh[l] = (const float*)d_in[21 + 4 * l];
    }
    const float* W0 = (const float*)d_in[30], *b0 = (const float*)d_in[31];
    const float* W1 = (const float*)d_in[32], *b1 = (const float*)d_in[33];
    const float* W2 = (const float*)d_in[34], *b2 = (const float*)d_in[35];

    // ---- workspace carve-up (256B aligned); ~430 MB
    char* wp = (char*)d_ws;
    auto carve = [&](size_t bytes) -> void* {
        void* p = wp; wp += (bytes + 255) & ~(size_t)255; return p;
    };
    _Float16* ew    = (_Float16*)carve((size_t)En * DN * DN * 2);  // 327.7 MB
    _Float16* ewh16 = (_Float16*)carve((size_t)En * DEH * 2);
    float*    h     = (float*)carve((size_t)Vn * DN * 4);
    _Float16* h16   = (_Float16*)carve((size_t)Vn * DN * 2);
    _Float16* m16   = (_Float16*)carve((size_t)Vn * DN * 2);
    float*    agg   = (float*)carve((size_t)Vn * DN * 4);
    float*    gi    = (float*)carve((size_t)Vn * 3 * DN * 4);
    float*    gh    = (float*)carve((size_t)Vn * 3 * DN * 4);
    float*    ep    = (float*)carve((size_t)En * DN * 4);
    float*    feat  = (float*)carve((size_t)Vn * 128 * 4);
    float*    esc   = (float*)carve((size_t)Vn * 4);
    int*      gst   = (int*)carve(Bg * 4);
    int*      gen   = (int*)carve(Bg * 4);
    // s2s f32 state: hs[3] | cs[3]
    float* s2s = (float*)carve((size_t)6 * Bg * DS2S * 4);
    float *hs[NLSTM], *cs[NLSTM];
    for (int l = 0; l < NLSTM; ++l) {
        hs[l] = s2s + l * Bg * DS2S;
        cs[l] = s2s + (NLSTM + l) * Bg * DS2S;
    }
    // s2s f16 mirrors: qstar16 | hs16[3] (contiguous for one zero pass)
    _Float16* s2sh = (_Float16*)carve((size_t)(Bg * 256 + 3 * Bg * DS2S) * 2);
    _Float16* qstar16 = s2sh;
    _Float16* hs16[NLSTM];
    for (int l = 0; l < NLSTM; ++l) hs16[l] = s2sh + Bg * 256 + l * Bg * DS2S;
    float* ga = (float*)carve((size_t)Bg * 4 * DS2S * 4);
    float* gb = (float*)carve((size_t)Bg * 4 * DS2S * 4);
    _Float16* x1h  = (_Float16*)carve((size_t)Bg * FFNHID * 2);
    _Float16* embh = (_Float16*)carve((size_t)Bg * FFNEMB * 2);
    // f16 weight prep (all N-major: B[n*K+k])
    _Float16* nf16  = (_Float16*)carve((size_t)Vn * NATF * 2);
    _Float16* ef16  = (_Float16*)carve((size_t)En * NBF * 2);
    _Float16* WpnT  = (_Float16*)carve((size_t)NATF * DN * 2);
    _Float16* We1T  = (_Float16*)carve((size_t)NBF * DEH * 2);
    _Float16* We2T  = (_Float16*)carve((size_t)DEH * DN * DN * 2);
    _Float16* WpeT  = (_Float16*)carve((size_t)NBF * DN * 2);
    _Float16* gwih16 = (_Float16*)carve((size_t)3 * DN * DN * 2);
    _Float16* gwhh16 = (_Float16*)carve((size_t)3 * DN * DN * 2);
    _Float16* lwih16[NLSTM], *lwhh16[NLSTM];
    for (int l = 0; l < NLSTM; ++l) {
        lwih16[l] = (_Float16*)carve((size_t)4 * DS2S * (l ? DS2S : FFNIN) * 2);
        lwhh16[l] = (_Float16*)carve((size_t)4 * DS2S * DS2S * 2);
    }
    _Float16* W0T = (_Float16*)carve((size_t)FFNIN * FFNHID * 2);
    _Float16* W1T = (_Float16*)carve((size_t)FFNHID * FFNEMB * 2);
    _Float16* W2T = (_Float16*)carve((size_t)FFNEMB * NTASK * 2);

    float* out    = (float*)d_out;
    float* logits = out + Bg * NTASK;
    float* emb    = out + 2 * Bg * NTASK;

    auto zero = [&](void* p, long nfloats) {
        k_zero<<<cdiv((int)nfloats, 256), 256, 0, stream>>>((float*)p, nfloats);
    };
    auto cvt = [&](const float* in, _Float16* o, long n) {
        k_cvt<<<cdiv((int)n, 256), 256, 0, stream>>>(in, o, n);
    };
    auto cvtT = [&](const float* in, _Float16* o, int K, int N) {
        k_cvtT<<<cdiv(K * N, 256), 256, 0, stream>>>(in, o, K, N);
    };
    auto ggrid = [&](int M, int N) { return dim3(cdiv(M, 16), cdiv(N, 128)); };

    // ---- weight / activation f16 prep ------------------------------------
    cvt(node_feats, nf16, (long)Vn * NATF);
    cvt(edge_feats, ef16, (long)En * NBF);
    cvtT(Wpn, WpnT, NATF, DN);
    cvtT(We1, We1T, NBF, DEH);
    cvtT(We2, We2T, DEH, DN * DN);
    cvtT(Wpe, WpeT, NBF, DN);
    cvt(gwih, gwih16, 3 * DN * DN);     // torch (out,in) already N-major
    cvt(gwhh, gwhh16, 3 * DN * DN);
    for (int l = 0; l < NLSTM; ++l) {
        cvt(lwih[l], lwih16[l], (long)4 * DS2S * (l ? DS2S : FFNIN));
        cvt(lwhh[l], lwhh16[l], (long)4 * DS2S * DS2S);
    }
    cvtT(W0, W0T, FFNIN, FFNHID);
    cvtT(W1, W1T, FFNHID, FFNEMB);
    cvtT(W2, W2T, FFNEMB, NTASK);

    // ---- input projections ------------------------------------------------
    // h0 = relu(node_feats @ Wpn + bpn)        (V,134)->(V,64)   [tail: K=134]
    k_gemm<true, false, true><<<ggrid(Vn, DN), 128, 0, stream>>>(
        nf16, WpnT, bpn, h, Vn, DN, NATF);
    cvt(h, h16, (long)Vn * DN);
    // ewh = relu(edge_feats @ We1 + be1)       (E,6)->(E,128)    [tail: K=6]
    k_gemm<true, true, true><<<ggrid(En, DEH), 128, 0, stream>>>(
        ef16, We1T, be1, ewh16, En, DEH, NBF);
    // ew = ewh @ We2 + be2  (f16 out)          (E,128)->(E,4096) [fast path]
    k_gemm<false, true, false><<<ggrid(En, DN * DN), 128, 0, stream>>>(
        ewh16, We2T, be2, ew, En, DN * DN, DEH);
    // e_proj = relu(edge_feats @ Wpe + bpe)    (E,6)->(E,64)     [tail: K=6]
    k_gemm<true, false, true><<<ggrid(En, DN), 128, 0, stream>>>(
        ef16, WpeT, bpe, ep, En, DN, NBF);

    // ---- 3 message-passing rounds ----------------------------------------
    for (int step = 0; step < NSTEPS; ++step) {
        zero(agg, (long)Vn * DN);
        k_msg_agg<<<cdiv(En, 8), 256, 0, stream>>>(h, ew, src, dst, agg);
        k_mix<<<cdiv(Vn * DN, 256), 256, 0, stream>>>(agg, nnb, h, m16);
        // gi = m @ gwih.T + gbih ; gh = h @ gwhh.T + gbhh   [fast path]
        k_gemm<false, false, false><<<ggrid(Vn, 3 * DN), 128, 0, stream>>>(
            m16, gwih16, gbih, gi, Vn, 3 * DN, DN);
        k_gemm<false, false, false><<<ggrid(Vn, 3 * DN), 128, 0, stream>>>(
            h16, gwhh16, gbhh, gh, Vn, 3 * DN, DN);
        k_gru<<<cdiv(Vn * DN, 256), 256, 0, stream>>>(gi, gh, h, h16);
    }

    // ---- readout features: feat[v] = sum_in [h_src ; e_proj] --------------
    zero(feat, (long)Vn * 128);
    k_feat<<<cdiv(En * 128, 256), 256, 0, stream>>>(h, ep, src, dst, feat);
    zero(gst, Bg);  zero(gen, Bg);
    k_bounds<<<cdiv(Vn, 256), 256, 0, stream>>>(gids, gst, gen);

    // ---- set2set ----------------------------------------------------------
    zero(s2s,  (long)6 * Bg * DS2S);                   // hs[], cs[] = 0
    zero(s2sh, (long)(Bg * 256 + 3 * Bg * DS2S) / 2);  // qstar16, hs16[] = 0
    for (int it = 0; it < NS2S; ++it) {
        const _Float16* x = qstar16;
        int kdim = FFNIN;                              // 256 for layer 0
        for (int l = 0; l < NLSTM; ++l) {
            k_gemm<false, false, false><<<ggrid(Bg, 4 * DS2S), 128, 0, stream>>>(
                x, lwih16[l], lbih[l], ga, Bg, 4 * DS2S, kdim);
            k_gemm<false, false, false><<<ggrid(Bg, 4 * DS2S), 128, 0, stream>>>(
                hs16[l], lwhh16[l], lbhh[l], gb, Bg, 4 * DS2S, DS2S);
            k_lstm<<<cdiv(Bg * DS2S, 256), 256, 0, stream>>>(ga, gb, hs[l], hs16[l], cs[l]);
            x = hs16[l];
            kdim = DS2S;
        }
        k_escore<<<cdiv(Vn, 8), 256, 0, stream>>>(feat, hs[NLSTM - 1], gids, esc);
        k_attn<<<Bg, 128, 0, stream>>>(feat, esc, hs[NLSTM - 1], gst, gen, qstar16);
    }

    // ---- FFN head ---------------------------------------------------------
    k_gemm<true, true, true><<<ggrid(Bg, FFNHID), 128, 0, stream>>>(
        qstar16, W0T, b0, x1h, Bg, FFNHID, FFNIN);           // N=300 tail
    k_gemm<true, false, true><<<ggrid(Bg, FFNEMB), 128, 0, stream>>>(
        x1h, W1T, b1, emb, Bg, FFNEMB, FFNHID);              // K=300 tail
    cvt(emb, embh, (long)Bg * FFNEMB);
    k_gemm<false, false, true><<<ggrid(Bg, NTASK), 128, 0, stream>>>(
        embh, W2T, b2, logits, Bg, NTASK, FFNEMB);           // N=138 tail
    k_sigmoid<<<cdiv(Bg * NTASK, 256), 256, 0, stream>>>(out);
}
